// AdditiveAttention_68521908241160
// MI455X (gfx1250) — matrix operand order repaired and checked by reading, not compiled, not run on hardware
//
#include <hip/hip_runtime.h>

#define B_SZ 32
#define L_SZ 4096
#define M_SZ 1024
#define A_SZ 256

typedef __attribute__((ext_vector_type(16))) __bf16 v16bf;
typedef __attribute__((ext_vector_type(8)))  float  v8f;

typedef __attribute__((ext_vector_type(4))) unsigned int tdm_v4u;
typedef __attribute__((ext_vector_type(8))) int          tdm_v8i;
typedef __attribute__((ext_vector_type(4))) int          tdm_v4i;

union AFrag { v16bf v; unsigned short h[16]; };
union BFrag { v16bf v; uint4 q[2]; };

#if __has_builtin(__builtin_amdgcn_tensor_load_to_lds) && __has_builtin(__builtin_amdgcn_s_wait_tensorcnt)
#define HAVE_TDM 1
#else
#define HAVE_TDM 0
#endif

#if defined(__has_include)
#if __has_include(<hip/amd_detail/amd_gfx1250_TDM.h>)
#define TDM_ARITY 6
#endif
#endif
#ifndef TDM_ARITY
#define TDM_ARITY 5
#endif

// LDS row stride for staged B: 64B of data + 16B pad = 80B (20 banks -> conflict-free b128 reads)
#define BSTRIDE_H 40                      // in ushorts
#define BBUF_BYTES (A_SZ * BSTRIDE_H * 2) // 20480 bytes per buffer

__device__ __forceinline__ unsigned short f2bf(float f) {
    unsigned int u = __float_as_uint(f);
    u += 0x7FFFu + ((u >> 16) & 1u);   // round-to-nearest-even
    return (unsigned short)(u >> 16);
}

#if HAVE_TDM
// TDM: load the 256(a) x 32(k) bf16 slice of u_W starting at column k0 into LDS,
// padded to an 80-byte row stride (pad 16B every 64B stored).
__device__ __forceinline__ void tdm_load_uW(const unsigned short* uWbf, int k0,
                                            unsigned lds_byte_off) {
    unsigned long long ga = (unsigned long long)(uintptr_t)uWbf + (unsigned)(k0 * 2);
    tdm_v4u g0;
    g0.x = 1u;                                     // count=1 (valid descriptor)
    g0.y = lds_byte_off;                           // lds_addr
    g0.z = (unsigned)ga;                           // global_addr[31:0]
    g0.w = (unsigned)(ga >> 32) | (2u << 30);      // global_addr[56:32] | type=2
    tdm_v8i g1;
    g1[0] = (int)((1u << 16)       // data_size = 1 (2 bytes)
                | (1u << 20)       // pad_enable
                | (3u << 22)       // pad_interval = 3 -> every 64 bytes
                | (3u << 25));     // pad_amount   = 3 -> 4 DWORDs = 16 bytes
    g1[1] = (int)(1024u << 16);    // tensor_dim0[15:0] = 1024 (row length, elements)
    g1[2] = (int)(256u << 16);     // tensor_dim1[15:0] = 256 rows
    g1[3] = (int)(32u << 16);      // tile_dim0 = 32 elements (64 bytes)
    g1[4] = 256;                   // tile_dim1 = 256 rows
    g1[5] = 1024;                  // tensor_dim0_stride = 1024 elements
    g1[6] = 0;
    g1[7] = 0;
    tdm_v4i z4 = {0, 0, 0, 0};
#if TDM_ARITY == 6
    tdm_v8i z8 = {0, 0, 0, 0, 0, 0, 0, 0};
    __builtin_amdgcn_tensor_load_to_lds(g0, g1, z4, z4, z8, 0);
#else
    __builtin_amdgcn_tensor_load_to_lds(g0, g1, z4, z4, 0);
#endif
}
#endif

// ---------------------------------------------------------------- prep: u_W f32 -> bf16
__global__ __launch_bounds__(256)
void cvt_uW_kernel(const float* __restrict__ uW, unsigned short* __restrict__ out, int n) {
    int i = blockIdx.x * 256 + threadIdx.x;
    if (i < n) out[i] = f2bf(uW[i]);
}

// ---------------------------------------------------------------- prep: wqb[b][a] = q.wW + w_b + u_b
__global__ __launch_bounds__(256)
void wqb_kernel(const float* __restrict__ query, const float* __restrict__ wW,
                const float* __restrict__ wb, const float* __restrict__ ub,
                float* __restrict__ wqb) {
    int b = blockIdx.x;
    int a = threadIdx.x;
    const float* qr = query + (size_t)b * M_SZ;
    const float* wr = wW + (size_t)a * M_SZ;
    float acc = 0.f;
    #pragma unroll 8
    for (int k = 0; k < M_SZ; ++k) acc += qr[k] * wr[k];
    wqb[b * A_SZ + a] = acc + wb[a] + ub[a];
}

// ---------------------------------------------------------------- main: scores via WMMA bf16
// WG = 8 waves covering 128 consecutive l-rows of one batch. Per K=32 step the
// 256x32 bf16 B slice of u_W is staged ONCE per WG into LDS (TDM, double-buffered),
// then all 8 waves build B fragments from LDS (conflict-free, 2-deep pipelined
// ds_load_b128) and run 16 WMMAs each against in-register bf16 A fragments.
__global__ __launch_bounds__(256)
void scores_kernel(const float* __restrict__ memory,
                   const unsigned short* __restrict__ uWbf,
                   const float* __restrict__ wqb,
                   const float* __restrict__ vW,
                   float* __restrict__ scores) {
    __shared__ unsigned short bstage[2][A_SZ * BSTRIDE_H];   // 2 x 20KB

    const int tid  = threadIdx.x;
    const int lane = tid & 31;
    const int wv   = tid >> 5;                 // wave in WG, 0..7
    const int b       = blockIdx.x >> 5;       // 32 WGs per batch
    const int tile128 = blockIdx.x & 31;
    const int l_base  = tile128 * 128 + wv * 16;
    const int lo = lane & 15;
    const int hi = lane >> 4;

    const float* rowp = memory + ((size_t)b * L_SZ + l_base + lo) * M_SZ;

    v8f acc[16];
    #pragma unroll
    for (int t = 0; t < 16; ++t) {
        #pragma unroll
        for (int r = 0; r < 8; ++r) acc[t][r] = 0.f;
    }

#if HAVE_TDM
    const unsigned lds_base = (unsigned)(uintptr_t)&bstage[0][0];
    if (wv == 0) tdm_load_uW(uWbf, 0, lds_base);            // prologue -> buf 0
#endif

    for (int ks = 0; ks < 32; ++ks) {
        const int k0  = ks * 32;
        const int cur = ks & 1;

        // Issue A-fragment global loads BEFORE the sync: they only depend on
        // `memory`, so HBM latency overlaps the barrier / TDM handoff.
        const int o = k0 + 16 * hi;
        const float4 f0 = *(const float4*)(rowp + o);
        const float4 f1 = *(const float4*)(rowp + o + 4);
        const float4 f2 = *(const float4*)(rowp + o + 8);
        const float4 f3 = *(const float4*)(rowp + o + 12);

#if HAVE_TDM
        if (wv == 0) __builtin_amdgcn_s_wait_tensorcnt(0);   // buf[cur] ready
        __syncthreads();                                     // publish cur; license other
        if (wv == 0 && ks < 31)
            tdm_load_uW(uWbf, k0 + 32, lds_base + (unsigned)(1 - cur) * BBUF_BYTES);
#else
        __syncthreads();
        {   // cooperative stage: thread t copies row a=t (64B) into padded LDS row
            const uint4* src = (const uint4*)(uWbf + (size_t)tid * M_SZ + k0);
            uint4* dst = (uint4*)&bstage[cur][tid * BSTRIDE_H];
            dst[0] = src[0]; dst[1] = src[1]; dst[2] = src[2]; dst[3] = src[3];
        }
        __syncthreads();
#endif

        // A fragment (contiguous K, matching B): lanes 0-15 hold K = k0+[0..15];
        // lanes 16-31 hold K = k0+16+[0..15]. f32 -> bf16 in-register.
        AFrag a;
        a.h[0]  = f2bf(f0.x); a.h[1]  = f2bf(f0.y); a.h[2]  = f2bf(f0.z); a.h[3]  = f2bf(f0.w);
        a.h[4]  = f2bf(f1.x); a.h[5]  = f2bf(f1.y); a.h[6]  = f2bf(f1.z); a.h[7]  = f2bf(f1.w);
        a.h[8]  = f2bf(f2.x); a.h[9]  = f2bf(f2.y); a.h[10] = f2bf(f2.z); a.h[11] = f2bf(f2.w);
        a.h[12] = f2bf(f3.x); a.h[13] = f2bf(f3.y); a.h[14] = f2bf(f3.z); a.h[15] = f2bf(f3.w);

        // B fragment: lane n holds 16 contiguous bf16 of u_W row (t*16+lo),
        // K offset 16*hi within the staged 32-wide slice. 2-deep software
        // pipeline so tile t+1's ds_load_b128s are in flight during WMMA t.
        const unsigned bbase = (unsigned)lo * BSTRIDE_H + 16u * hi;
        BFrag bfr[2];
        {
            const uint4* bp = (const uint4*)&bstage[cur][bbase];
            bfr[0].q[0] = bp[0];
            bfr[0].q[1] = bp[1];
        }
        #pragma unroll
        for (int t = 0; t < 16; ++t) {
            if (t < 15) {
                const uint4* bp =
                    (const uint4*)&bstage[cur][bbase + (unsigned)(t + 1) * 16u * BSTRIDE_H];
                bfr[(t + 1) & 1].q[0] = bp[0];
                bfr[(t + 1) & 1].q[1] = bp[1];
            }
            acc[t] = __builtin_amdgcn_wmma_f32_16x16x32_bf16(
                false, a.v, false, bfr[t & 1].v, (short)0, acc[t], false, false);
        }
    }

    // epilogue: scores[l] = sum_a vW[a] * tanh(wqb[b][a] + um[l][a])
    float sc[8];
    #pragma unroll
    for (int r = 0; r < 8; ++r) sc[r] = 0.f;
    #pragma unroll
    for (int t = 0; t < 16; ++t) {
        const int aidx = t * 16 + lo;              // D layout: N = lane&15
        const float bias = wqb[b * A_SZ + aidx];
        const float va   = vW[aidx];
        #pragma unroll
        for (int r = 0; r < 8; ++r)
            sc[r] += va * tanhf(bias + acc[t][r]);
    }
    // reduce over N (16 lanes per half); D layout: lanes 0-15 -> rows r, 16-31 -> rows r+8
    #pragma unroll
    for (int m = 1; m < 16; m <<= 1) {
        #pragma unroll
        for (int r = 0; r < 8; ++r) sc[r] += __shfl_xor(sc[r], m, 32);
    }
    if (lo == 0) {
        float* out = scores + (size_t)b * L_SZ + l_base + 8 * hi;
        #pragma unroll
        for (int r = 0; r < 8; ++r) out[r] = sc[r];
    }
}

// ---------------------------------------------------------------- softmax over L per batch row
__global__ __launch_bounds__(256)
void softmax_kernel(const float* __restrict__ scores, const float* __restrict__ mask,
                    float* __restrict__ weights) {
    __shared__ float red[256];
    const int b = blockIdx.x, tid = threadIdx.x;
    float s[16];
    float mx = -INFINITY;
    #pragma unroll
    for (int i = 0; i < 16; ++i) {
        int l = tid + i * 256;
        float v = scores[b * L_SZ + l] + __logf(mask[b * L_SZ + l]);
        s[i] = v;
        mx = fmaxf(mx, v);
    }
    red[tid] = mx; __syncthreads();
    for (int off = 128; off > 0; off >>= 1) {
        if (tid < off) red[tid] = fmaxf(red[tid], red[tid + off]);
        __syncthreads();
    }
    mx = red[0]; __syncthreads();
    float sum = 0.f;
    #pragma unroll
    for (int i = 0; i < 16; ++i) { s[i] = __expf(s[i] - mx); sum += s[i]; }
    red[tid] = sum; __syncthreads();
    for (int off = 128; off > 0; off >>= 1) {
        if (tid < off) red[tid] += red[tid + off];
        __syncthreads();
    }
    const float inv = 1.f / red[0];
    #pragma unroll
    for (int i = 0; i < 16; ++i)
        weights[b * L_SZ + tid + i * 256] = s[i] * inv;
}

// ---------------------------------------------------------------- aggr[b][m] = sum_l w[l]*memory[b][l][m]
__global__ __launch_bounds__(256)
void aggr_kernel(const float* __restrict__ memory, const float* __restrict__ weights,
                 float* __restrict__ aggr) {
    __shared__ float wls[256];
    const int bz = blockIdx.x;            // B * 4 mchunks * 8 lchunks = 1024
    const int b = bz >> 5;
    const int mchunk = (bz >> 3) & 3;
    const int lchunk = bz & 7;
    const int m = mchunk * 256 + threadIdx.x;
    const float* memb = memory + (size_t)b * L_SZ * M_SZ;
    const float* wb   = weights + (size_t)b * L_SZ;
    float acc = 0.f;
    for (int lt = 0; lt < 2; ++lt) {
        const int l0 = lchunk * 512 + lt * 256;
        __syncthreads();
        wls[threadIdx.x] = wb[l0 + threadIdx.x];
        __syncthreads();
        #pragma unroll 8
        for (int i = 0; i < 256; ++i)
            acc += wls[i] * memb[(size_t)(l0 + i) * M_SZ + m];
    }
    unsafeAtomicAdd(&aggr[b * M_SZ + m], acc);   // -> global_atomic_add_f32
}

// ---------------------------------------------------------------- launcher
extern "C" void kernel_launch(void* const* d_in, const int* in_sizes, int n_in,
                              void* d_out, int out_size, void* d_ws, size_t ws_size,
                              hipStream_t stream) {
    const float* query  = (const float*)d_in[0];
    const float* memory = (const float*)d_in[1];
    const float* mask   = (const float*)d_in[2];
    const float* w_W    = (const float*)d_in[3];
    const float* w_b    = (const float*)d_in[4];
    const float* u_W    = (const float*)d_in[5];
    const float* u_b    = (const float*)d_in[6];
    const float* v_W    = (const float*)d_in[7];
    // d_in[8] = v_b : softmax-invariant, not needed for the outputs

    char* ws = (char*)d_ws;
    unsigned short* uWbf = (unsigned short*)ws;                               // 512 KB
    float* wqb    = (float*)(ws + (size_t)A_SZ * M_SZ * 2);                   // 32 KB
    float* scores = (float*)(ws + (size_t)A_SZ * M_SZ * 2 + B_SZ * A_SZ * 4); // 512 KB

    float* aggr    = (float*)d_out;                 // [B, M]
    float* weights = (float*)d_out + B_SZ * M_SZ;   // [B, L]

    cvt_uW_kernel<<<(A_SZ * M_SZ + 255) / 256, 256, 0, stream>>>(u_W, uWbf, A_SZ * M_SZ);
    wqb_kernel<<<B_SZ, 256, 0, stream>>>(query, w_W, w_b, u_b, wqb);
    scores_kernel<<<B_SZ * (L_SZ / 128), 256, 0, stream>>>(memory, uWbf, wqb, v_W, scores);
    softmax_kernel<<<B_SZ, 256, 0, stream>>>(scores, mask, weights);
    hipMemsetAsync(aggr, 0, (size_t)B_SZ * M_SZ * sizeof(float), stream);
    aggr_kernel<<<B_SZ * 4 * 8, 256, 0, stream>>>(memory, weights, aggr);
}
